// LSTMAugmented_84473416778470
// MI455X (gfx1250) — matrix operand order repaired and checked by reading, not compiled, not run on hardware
//
#include <hip/hip_runtime.h>
#include <stdint.h>

#define B_    8
#define T_    2048
#define I_    256
#define H_    256
#define ROWS  (B_ * T_)     // 16384
#define SIXH  (6 * H_)      // 1536
#define FIVEH (5 * H_)      // 1280

typedef __attribute__((ext_vector_type(16))) __bf16 v16bf;
typedef __attribute__((ext_vector_type(8)))  float  v8f;

struct Pack32 { uint4 lo, hi; };

__device__ __forceinline__ v16bf frag_from(uint4 lo, uint4 hi) {
  Pack32 p{lo, hi};
  return __builtin_bit_cast(v16bf, p);
}

// A-matrix 16x32 bf16 fragment (ISA 7.12.2): lane holds row M = lane%16;
// regs 0-3 = 8 contiguous bf16 at k0, regs 4-7 = 8 contiguous at k0+16,
// where k0 = 32*kk + 8*(lane/16).
__device__ __forceinline__ v16bf load_frag_a(const unsigned short* row, int kk, int lane) {
  int k0 = kk * 32 + ((lane >> 4) << 3);
  const uint4* p = reinterpret_cast<const uint4*>(row + k0);
  return frag_from(p[0], p[2]);
}

// B-matrix 32x16 bf16 fragment: lane holds column N = lane%16;
// regs 0-3 = 8 contiguous bf16 at k0, regs 4-7 = 8 contiguous at k0+16,
// where k0 = 32*kk + 8*(lane/16) of the row-major W row (same K order as A).
__device__ __forceinline__ v16bf load_frag_b(const unsigned short* row, int kk, int lane) {
  int k0 = kk * 32 + ((lane >> 4) << 3);
  const uint4* p = reinterpret_cast<const uint4*>(row + k0);
  return frag_from(p[0], p[2]);
}

__device__ __forceinline__ unsigned short f2bf(float f) {
  unsigned int u = __float_as_uint(f);
  u += 0x7FFFu + ((u >> 16) & 1u);   // round-to-nearest-even
  return (unsigned short)(u >> 16);
}

__device__ __forceinline__ float sigm(float x) {
  return 1.0f / (1.0f + __expf(-x));
}
__device__ __forceinline__ float tanh_(float x) {
  x = fminf(15.0f, fmaxf(-15.0f, x));
  float e = __expf(2.0f * x);
  return (e - 1.0f) / (e + 1.0f);
}

#define WMMA_BF16(a, b, c) \
  __builtin_amdgcn_wmma_f32_16x16x32_bf16(false, (a), false, (b), (short)0, (c), false, false)

// ---------------------------------------------------------------- utilities
__global__ void k_zero(float* __restrict__ p, long long n) {
  long long i = (long long)blockIdx.x * blockDim.x + threadIdx.x;
  long long s = (long long)gridDim.x * blockDim.x;
  for (; i < n; i += s) p[i] = 0.0f;
}

// Build bf16 copies of x (forward order) and x reversed-within-length.
__global__ void k_prep_x(const float* __restrict__ x, const int* __restrict__ lengths,
                         unsigned short* __restrict__ xf, unsigned short* __restrict__ xr) {
  int r = blockIdx.x;            // row in [0, B*T)
  int i = threadIdx.x;           // feature in [0, I)
  int b = r >> 11;               // T = 2048
  int t = r & (T_ - 1);
  size_t o = (size_t)r * I_ + i;
  xf[o] = f2bf(x[o]);
  int idx = lengths[b] - 1 - t;
  float vr = (idx >= 0) ? x[((size_t)b * T_ + idx) * I_ + i] : 0.0f;
  xr[o] = f2bf(vr);
}

__global__ void k_prep_w(const float* __restrict__ Wi_f, const float* __restrict__ Wi_b,
                         const float* __restrict__ Ws_f, const float* __restrict__ Ws_b,
                         unsigned short* __restrict__ Wibf, unsigned short* __restrict__ Wsbf) {
  int i = blockIdx.x * blockDim.x + threadIdx.x;
  int s = gridDim.x * blockDim.x;
  const int NWI = SIXH * I_;     // 393216
  const int NWS = FIVEH * H_;    // 327680
  for (int k = i; k < NWI; k += s) { Wibf[k] = f2bf(Wi_f[k]); Wibf[NWI + k] = f2bf(Wi_b[k]); }
  for (int k = i; k < NWS; k += s) { Wsbf[k] = f2bf(Ws_f[k]); Wsbf[NWS + k] = f2bf(Ws_b[k]); }
}

// ---------------------------------------------------------------- input-projection GEMM
// xp[dir] = x(dir) @ Wi(dir)^T : [16384,256] x [256,1536] -> f32
// block = 8 waves; wave w owns M-tile (blockIdx.x*8+w), 64 output cols (blockIdx.y*64).
// kk-outer, 4 independent accumulator chains; A fragment reused across the 4 tiles.
__global__ __launch_bounds__(256) void
k_gemm(const unsigned short* __restrict__ xf, const unsigned short* __restrict__ xr,
       const unsigned short* __restrict__ Wibf, float* __restrict__ xp) {
  int wave = threadIdx.x >> 5, lane = threadIdx.x & 31;
  int mtile = blockIdx.x * 8 + wave;
  int n0    = blockIdx.y * 64;
  int dir   = blockIdx.z;
  const unsigned short* A = dir ? xr : xf;
  const unsigned short* W = Wibf + (size_t)dir * SIXH * I_;
  float* O = xp + (size_t)dir * ROWS * SIXH;

  int lm = lane & 15;
  const unsigned short* arow = A + (size_t)(mtile * 16 + lm) * I_;
  const unsigned short* w0 = W + (size_t)(n0 + 0 * 16 + lm) * I_;
  const unsigned short* w1 = W + (size_t)(n0 + 1 * 16 + lm) * I_;
  const unsigned short* w2 = W + (size_t)(n0 + 2 * 16 + lm) * I_;
  const unsigned short* w3 = W + (size_t)(n0 + 3 * 16 + lm) * I_;

  v8f acc0 = {}, acc1 = {}, acc2 = {}, acc3 = {};
#pragma unroll
  for (int kk = 0; kk < 8; ++kk) {
    v16bf a  = load_frag_a(arow, kk, lane);
    v16bf b0 = load_frag_b(w0, kk, lane);
    v16bf b1 = load_frag_b(w1, kk, lane);
    v16bf b2 = load_frag_b(w2, kk, lane);
    v16bf b3 = load_frag_b(w3, kk, lane);
    acc0 = WMMA_BF16(a, b0, acc0);
    acc1 = WMMA_BF16(a, b1, acc1);
    acc2 = WMMA_BF16(a, b2, acc2);
    acc3 = WMMA_BF16(a, b3, acc3);
  }

  int m0 = mtile * 16 + (lane >> 4) * 8;   // C/D: lane-half selects M 0..7 vs 8..15
#pragma unroll
  for (int r = 0; r < 8; ++r) {
    O[(size_t)(m0 + r) * SIXH + (n0 + 0 * 16 + lm)] = acc0[r];
    O[(size_t)(m0 + r) * SIXH + (n0 + 1 * 16 + lm)] = acc1[r];
    O[(size_t)(m0 + r) * SIXH + (n0 + 2 * 16 + lm)] = acc2[r];
    O[(size_t)(m0 + r) * SIXH + (n0 + 3 * 16 + lm)] = acc3[r];
  }
}

// ---------------------------------------------------------------- recurrent scan
// One workgroup per direction; 20 waves, each wave owns 4 N-tiles of Ws^T with the
// bf16 B-fragments held register-resident for all 2048 steps.
__global__ __launch_bounds__(640) void
k_scan(const float* __restrict__ xp, const float* __restrict__ bs_f,
       const float* __restrict__ bs_b, const int* __restrict__ lengths,
       const unsigned short* __restrict__ Wsbf, float* __restrict__ out) {
  __shared__ unsigned short hbf[16 * H_];   // h as bf16, rows 8..15 stay zero (A padding)
  __shared__ float cf[B_ * H_];             // cell state, f32
  __shared__ float ps[B_ * FIVEH];          // h @ Ws^T for this step
  __shared__ float bsl[FIVEH];
  __shared__ int   lenl[B_];

  int tid = threadIdx.x;
  int dir = blockIdx.x;
  int wv = tid >> 5, lane = tid & 31;
  int lm = lane & 15;

  const float* bs = dir ? bs_b : bs_f;
  for (int i = tid; i < 16 * H_; i += 640) hbf[i] = 0;
  for (int i = tid; i < B_ * H_; i += 640) cf[i] = 0.0f;
  for (int i = tid; i < FIVEH; i += 640) bsl[i] = bs[i];
  if (tid < B_) lenl[tid] = lengths[tid];

  // Register-resident recurrent weights: 4 tiles x 8 k-steps of v16bf per lane.
  v16bf Bf[4][8];
#pragma unroll
  for (int j = 0; j < 4; ++j) {
    int ocol = (wv * 4 + j) * 16 + lm;
    const unsigned short* wrow = Wsbf + ((size_t)dir * FIVEH + ocol) * H_;
#pragma unroll
    for (int kk = 0; kk < 8; ++kk) Bf[j][kk] = load_frag_b(wrow, kk, lane);
  }
  __syncthreads();

  const float* XP = xp + (size_t)dir * ROWS * SIXH;
  const unsigned short* hrow = hbf + lm * H_;

  for (int t = 0; t < T_; ++t) {
    // ps = h @ Ws^T  (M=16 padded, rows 8..15 zero)
    v8f a0 = {}, a1 = {}, a2 = {}, a3 = {};
#pragma unroll
    for (int kk = 0; kk < 8; ++kk) {
      v16bf af = load_frag_a(hrow, kk, lane);
      a0 = WMMA_BF16(af, Bf[0][kk], a0);
      a1 = WMMA_BF16(af, Bf[1][kk], a1);
      a2 = WMMA_BF16(af, Bf[2][kk], a2);
      a3 = WMMA_BF16(af, Bf[3][kk], a3);
    }
    if (lane < 16) {   // lanes 0-15 hold valid batch rows 0..7 in regs 0..7
#pragma unroll
      for (int r = 0; r < 8; ++r) {
        ps[r * FIVEH + (wv * 4 + 0) * 16 + lane] = a0[r];
        ps[r * FIVEH + (wv * 4 + 1) * 16 + lane] = a1[r];
        ps[r * FIVEH + (wv * 4 + 2) * 16 + lane] = a2[r];
        ps[r * FIVEH + (wv * 4 + 3) * 16 + lane] = a3[r];
      }
    }
    __syncthreads();

    // gates + state update: 2048 (b,hcol) elements over 640 threads
    for (int e = tid; e < B_ * H_; e += 640) {
      int b = e >> 8, hc = e & (H_ - 1);
      size_t base = ((size_t)b * T_ + t) * SIXH;
      const float* pr = ps + b * FIVEH;
      float s0  = XP[base + hc]           + pr[hc]           + bsl[hc];
      float s1  = XP[base + H_ + hc]      + pr[H_ + hc]      + bsl[H_ + hc];
      float s2  = XP[base + 2 * H_ + hc]  + pr[2 * H_ + hc]  + bsl[2 * H_ + hc];
      float s3  = XP[base + 3 * H_ + hc]  + pr[3 * H_ + hc]  + bsl[3 * H_ + hc];
      float s4  = XP[base + 4 * H_ + hc]  + pr[4 * H_ + hc]  + bsl[4 * H_ + hc];
      float pin = XP[base + 5 * H_ + hc];
      float ig = sigm(s0), fg = sigm(s1), mi = tanh_(s2), og = sigm(s3), hg = sigm(s4);
      float c_old = cf[e];
      float c_new = ig * mi + fg * c_old;
      float ov = og * tanh_(c_new);
      ov = hg * ov + (1.0f - hg) * pin;
      int L = lenl[b];
      bool msk = t < L;
      cf[e] = msk ? c_new : c_old;
      if (msk) {
        hbf[b * H_ + hc] = f2bf(ov);
        int tr = dir ? (L - 1 - t) : t;           // backward writes reversed in place
        out[((size_t)b * T_ + tr) * (2 * H_) + (dir << 8) + hc] = ov;
      }
      if (t + 1 < T_) __builtin_prefetch(XP + base + SIXH + hc, 0, 1);
    }
    __syncthreads();
  }
}

// ---------------------------------------------------------------- launch
extern "C" void kernel_launch(void* const* d_in, const int* in_sizes, int n_in,
                              void* d_out, int out_size, void* d_ws, size_t ws_size,
                              hipStream_t stream) {
  (void)in_sizes; (void)n_in; (void)out_size; (void)ws_size;
  const float* x       = (const float*)d_in[0];
  const int*   lengths = (const int*)d_in[1];
  const float* Wi_f    = (const float*)d_in[2];
  const float* Ws_f    = (const float*)d_in[3];
  const float* bs_f    = (const float*)d_in[4];
  const float* Wi_b    = (const float*)d_in[5];
  const float* Ws_b    = (const float*)d_in[6];
  const float* bs_b    = (const float*)d_in[7];
  float* out = (float*)d_out;

  char* ws = (char*)d_ws;
  size_t off = 0;
  float* xp = (float*)(ws + off);          off += (size_t)2 * ROWS * SIXH * sizeof(float);   // 192 MiB
  unsigned short* xf   = (unsigned short*)(ws + off); off += (size_t)ROWS * I_ * 2;           // 8 MiB
  unsigned short* xr   = (unsigned short*)(ws + off); off += (size_t)ROWS * I_ * 2;           // 8 MiB
  unsigned short* Wibf = (unsigned short*)(ws + off); off += (size_t)2 * SIXH * I_ * 2;       // 1.5 MiB
  unsigned short* Wsbf = (unsigned short*)(ws + off); off += (size_t)2 * FIVEH * H_ * 2;      // 1.25 MiB

  k_zero<<<1024, 256, 0, stream>>>(out, (long long)ROWS * 2 * H_);
  k_prep_x<<<ROWS, 256, 0, stream>>>(x, lengths, xf, xr);
  k_prep_w<<<512, 256, 0, stream>>>(Wi_f, Wi_b, Ws_f, Ws_b, Wibf, Wsbf);
  k_gemm<<<dim3(128, 24, 2), 256, 0, stream>>>(xf, xr, Wibf, xp);
  k_scan<<<dim3(2), 640, 0, stream>>>(xp, bs_f, bs_b, lengths, Wsbf, out);
}
